// DownstreamExpert_83545703842159
// MI455X (gfx1250) — matrix operand order repaired and checked by reading, not compiled, not run on hardware
//
#include <hip/hip_runtime.h>
#include <hip/hip_bf16.h>

typedef __attribute__((ext_vector_type(16))) _Float16 v16h;
typedef __attribute__((ext_vector_type(8)))  _Float16 v8h;
typedef __attribute__((ext_vector_type(8)))  float    v8f;

union V16H { v16h v; v8h h2[2]; _Float16 e[16]; };

#define SDTW_GAMMA 0.1f
#define SDTW_INVG  10.0f
#define SDTW_BIG   1e8f
#define SKEW       16          // diagonals per barrier interval / wave skew
#define RING       (4 * SKEW)  // ring slots per wave (>=2 intervals of slack)

// Offset of anti-diagonal d in the diagonal-major packing of an N x M matrix
// (N >= M).  Diagonal d holds cells (i, d-i), i in [max(0,d-M+1), min(d,N-1)],
// stored contiguously; total size is exactly N*M.
__device__ __forceinline__ int diag_base(int d, int N, int M) {
  if (d <= M) return d * (d + 1) / 2;
  if (d <= N) return M * (M + 1) / 2 + (d - M) * M;
  const int u = d - N;
  return M * (M + 1) / 2 + (N - M) * M + u * (M - 1) - u * (u - 1) / 2;
}

// ---------------------------------------------------------------------------
// Kernel 1: Z = A[rows,768] @ W[768,128] + b, then L2-normalize each row,
// emit f16 [rows,128].  Block = 256 threads = 8 waves; block owns 16 rows,
// wave w owns output columns [16w,16w+16).  K reduced in 24 WMMA steps.
// ---------------------------------------------------------------------------
__global__ __launch_bounds__(256) void proj_norm_kernel(
    const float* __restrict__ A, const float* __restrict__ W,
    const float* __restrict__ bias, _Float16* __restrict__ out)
{
  const int row0 = blockIdx.x * 16;
  const int lane = threadIdx.x & 31;
  const int wave = threadIdx.x >> 5;
  const int n0   = wave * 16;
  const int li   = lane & 15;        // A: M-row / C: N-col within tile
  const int kb   = (lane >> 4) * 8;  // A-frag half-lane K offset
  const int kbB  = (lane >> 4) * 16; // unused; B-frag also uses kb

  const float* arow = A + (size_t)(row0 + li) * 768;
  const float* wcol = W + (n0 + li);

  v8f c = {0.f, 0.f, 0.f, 0.f, 0.f, 0.f, 0.f, 0.f};
  for (int k0 = 0; k0 < 768; k0 += 32) {
    V16H a, b;
#pragma unroll
    for (int h = 0; h < 8; ++h) {
      a.e[h]     = (_Float16)arow[k0 + kb + h];
      a.e[h + 8] = (_Float16)arow[k0 + 16 + kb + h];
    }
#pragma unroll
    for (int h = 0; h < 8; ++h) {
      b.e[h]     = (_Float16)wcol[(size_t)(k0 + kb + h) * 128];
      b.e[h + 8] = (_Float16)wcol[(size_t)(k0 + 16 + kb + h) * 128];
    }
    c = __builtin_amdgcn_wmma_f32_16x16x32_f16(false, a.v, false, b.v,
                                               (short)0, c, false, false);
  }

  // bias, then per-row sum of squares: this wave holds 16 cols of rows
  // m = (lane>>4)*8 + r; reduce across the 16 lanes of each half-wave.
  const float bn = bias[n0 + li];
  float ss[8];
#pragma unroll
  for (int r = 0; r < 8; ++r) { float v = c[r] + bn; c[r] = v; ss[r] = v * v; }
#pragma unroll
  for (int mask = 1; mask < 16; mask <<= 1) {
#pragma unroll
    for (int r = 0; r < 8; ++r) ss[r] += __shfl_xor(ss[r], mask, 32);
  }

  __shared__ float part[8][16];
  __shared__ float invn[16];
  const int mh = (lane >> 4) * 8;
  if (li == 0) {
#pragma unroll
    for (int r = 0; r < 8; ++r) part[wave][mh + r] = ss[r];
  }
  __syncthreads();
  if (threadIdx.x < 16) {
    float t = 0.f;
#pragma unroll
    for (int w = 0; w < 8; ++w) t += part[w][threadIdx.x];
    invn[threadIdx.x] = rsqrtf(fmaxf(t, 1e-24f));  // == 1/max(||x||,1e-12)
  }
  __syncthreads();
#pragma unroll
  for (int r = 0; r < 8; ++r) {
    const int m = mh + r;
    out[(size_t)(row0 + m) * 128 + n0 + li] = (_Float16)(c[r] * invn[m]);
  }
}

// ---------------------------------------------------------------------------
// Kernel 2: D[i,j] = 2 - 2*dot(X_i, Y_j) (unit rows => squared L2 dist),
// written DIAGONAL-MAJOR so the DP scan reads contiguously.
// K = 128 => 4 WMMA steps.  Block = 8 waves: 16(M) x 128(N) output strip
// staged through LDS, then emitted grouped by anti-diagonal (runs of <=16).
// Grid: (N/16, M/128, B).
// ---------------------------------------------------------------------------
__global__ __launch_bounds__(256) void gram_dist_kernel(
    const _Float16* __restrict__ X, const _Float16* __restrict__ Y,
    float* __restrict__ Dd, int N, int M)
{
  __shared__ float tile[16][130];  // 16 x 128 strip (+pad)

  const int bat  = blockIdx.z;
  const int i0   = blockIdx.x * 16;
  const int lane = threadIdx.x & 31;
  const int wave = threadIdx.x >> 5;
  const int j0   = blockIdx.y * 128 + wave * 16;
  const int li   = lane & 15;
  const int kb   = (lane >> 4) * 8;
  const int kbB  = (lane >> 4) * 16;

  const _Float16* xr = X + ((size_t)bat * N + i0 + li) * 128;
  const _Float16* yr = Y + ((size_t)bat * M + j0 + li) * 128;

  v8f c = {0.f, 0.f, 0.f, 0.f, 0.f, 0.f, 0.f, 0.f};
#pragma unroll
  for (int k0 = 0; k0 < 128; k0 += 32) {
    V16H a, b;
    a.h2[0] = *(const v8h*)(xr + k0 + kb);        // K = k0+kb   .. +7
    a.h2[1] = *(const v8h*)(xr + k0 + 16 + kb);   // K = k0+16+kb.. +7
    b.h2[0] = *(const v8h*)(yr + k0 + kb);        // K = k0+kb   .. +7
    b.h2[1] = *(const v8h*)(yr + k0 + 16 + kb);   // K = k0+16+kb.. +7
    c = __builtin_amdgcn_wmma_f32_16x16x32_f16(false, a.v, false, b.v,
                                               (short)0, c, false, false);
  }
  // C layout: row m = kb + r, col = li  ->  stage strip in LDS
#pragma unroll
  for (int r = 0; r < 8; ++r)
    tile[kb + r][wave * 16 + li] = 2.0f - 2.0f * c[r];
  __syncthreads();

  // Emit diagonal-major.  Strip covers local diagonals s = 0..142
  // (global d = i0 + blockIdx.y*128 + s); each has <=16 cells, contiguous
  // in the packed layout (consecutive i <-> consecutive address).
  float* Db = Dd + (size_t)bat * N * M;
  const int j0s = blockIdx.y * 128;
  const int d0  = i0 + j0s;
  for (int e = threadIdx.x; e < 143 * 16; e += 256) {
    const int s   = e >> 4;
    const int k   = e & 15;
    const int ilo = (s > 127) ? (s - 127) : 0;
    const int i_l = ilo + k;
    if (i_l > 15 || i_l > s) continue;
    const int j_l = s - i_l;
    const int d   = d0 + s;
    int imin = d - M + 1; if (imin < 0) imin = 0;
    Db[diag_base(d, N, M) + (i0 + i_l) - imin] = tile[i_l][j_l];
  }
}

// ---------------------------------------------------------------------------
// Kernel 3: soft-DTW DP with SKEWED WAVES.  One 1024-thread block per
// (matrix-type, batch); lane l of wave w owns row i = 32w + l, with state
// p1 = v(i,d-1), p2 = v(i,d-2) in registers.  The i-1 neighbor comes from
// __shfl_up (wave32-synchronous, no barrier); only lane 0 reads wave w-1's
// lane-31 boundary from an LDS ring.  Wave w runs SKEW diagonals behind
// wave w-1, so one __syncthreads per SKEW steps suffices (~13x fewer
// barriers than one-per-diagonal).  D reads are coalesced (diag-major) and
// prefetched one diagonal ahead.
// ---------------------------------------------------------------------------
__global__ __launch_bounds__(1024) void sdtw_kernel(
    const float* __restrict__ Dxy, const float* __restrict__ Dxx,
    const float* __restrict__ Dyy, float* __restrict__ res)
{
  __shared__ float ring[32][RING];
  const int type = blockIdx.x;   // 0: xy, 1: xx, 2: yy
  const int bat  = blockIdx.y;
  const float* Db; int N, M;
  if (type == 0)      { Db = Dxy + (size_t)bat * 1024 * 768;  N = 1024; M = 768;  }
  else if (type == 1) { Db = Dxx + (size_t)bat * 1024 * 1024; N = 1024; M = 1024; }
  else                { Db = Dyy + (size_t)bat * 768 * 768;   N = 768;  M = 768;  }

  const int tid  = threadIdx.x;
  const int lane = tid & 31;
  const int w    = tid >> 5;
  const int i    = tid;             // row owned by this lane
  const int steps = N + M - 1;
  const int nw    = N >> 5;         // active waves
  const int T     = steps + SKEW * (nw - 1);
  const int dofs  = SKEW * w;

  float p1 = SDTW_BIG;   // v(i, d-1)
  float p2 = SDTW_BIG;   // v(i, d-2)
  float dcur = 0.f;      // D[i, d-i] for the current diagonal (prefetched)

  for (int t0 = 0; t0 < T; t0 += SKEW) {
#pragma unroll 4
    for (int s = 0; s < SKEW; ++s) {
      const int t = t0 + s;
      const int d = t - dofs;
      if (w < nw && d >= 0 && d < steps) {
        if (d == 0 && i == 0) dcur = Db[0];  // bootstrap the pipeline
        // issue next diagonal's load before this step's compute
        float dnxt = 0.f;
        {
          const int dn = d + 1;
          const int jn = dn - i;
          if (dn < steps && jn >= 0 && jn < M) {
            int imin = dn - M + 1; if (imin < 0) imin = 0;
            dnxt = Db[diag_base(dn, N, M) + i - imin];
          }
        }
        // neighbor (row i-1) values: in-wave shuffle; lane 0 from LDS ring
        float u1 = __shfl_up(p1, 1, 32);   // v(i-1, d-1)
        float u2 = __shfl_up(p2, 1, 32);   // v(i-1, d-2)
        if (lane == 0) {
          if (w == 0) {                    // virtual row -1: R[-1,-1] = 0
            u1 = SDTW_BIG;
            u2 = (d == 0) ? 0.f : SDTW_BIG;
          } else {
            u1 = (d >= 1) ? ring[w - 1][(t - SKEW - 1) & (RING - 1)] : SDTW_BIG;
            u2 = (d >= 2) ? ring[w - 1][(t - SKEW - 2) & (RING - 1)] : SDTW_BIG;
          }
        }
        const int j = d - i;
        float val = SDTW_BIG;
        if (j >= 0 && j < M) {
          const float m = fminf(u2, fminf(u1, p1));
          const float sm = __expf((m - u2) * SDTW_INVG) +
                           __expf((m - u1) * SDTW_INVG) +
                           __expf((m - p1) * SDTW_INVG);
          val = dcur + (m - SDTW_GAMMA * __logf(sm));
        }
        p2 = p1; p1 = val;
        if (lane == 31) ring[w][t & (RING - 1)] = val;  // publish boundary
        if (i == N - 1 && d == steps - 1) res[type * 16 + bat] = val;
        dcur = dnxt;
      }
    }
    __syncthreads();   // orders ring producer/consumer across intervals
  }
}

// ---------------------------------------------------------------------------
// Kernel 4: loss = mean_b(xy - 0.5*(xx+yy)) / (T1+T2)
// ---------------------------------------------------------------------------
__global__ void finish_kernel(const float* __restrict__ res, float* __restrict__ out)
{
  if (threadIdx.x == 0 && blockIdx.x == 0) {
    float s = 0.f;
    for (int b = 0; b < 16; ++b)
      s += res[b] - 0.5f * (res[16 + b] + res[32 + b]);
    out[0] = s * (1.0f / 16.0f) * (1.0f / 1792.0f);
  }
}

// ---------------------------------------------------------------------------
// Workspace layout (bytes)
// ---------------------------------------------------------------------------
static const size_t OFF_X   = 0;           // 16*1024*128 f16
static const size_t OFF_Y   = 4194304;     // 16*768*128  f16
static const size_t OFF_DXY = 7340032;     // 16*1024*768 f32  (diag-major)
static const size_t OFF_DXX = 57671680;    // 16*1024*1024 f32 (diag-major)
static const size_t OFF_DYY = 124780544;   // 16*768*768  f32  (diag-major)
static const size_t OFF_RES = 162529280;   // 48 f32

extern "C" void kernel_launch(void* const* d_in, const int* in_sizes, int n_in,
                              void* d_out, int out_size, void* d_ws, size_t ws_size,
                              hipStream_t stream) {
  (void)in_sizes; (void)n_in; (void)out_size; (void)ws_size;
  const float* feats = (const float*)d_in[0];  // [16,1024,768]
  const float* tgts  = (const float*)d_in[1];  // [16,768,768]
  const float* W     = (const float*)d_in[2];  // [768,128]
  const float* bias  = (const float*)d_in[3];  // [128]

  char* ws = (char*)d_ws;
  _Float16* Xh  = (_Float16*)(ws + OFF_X);
  _Float16* Yh  = (_Float16*)(ws + OFF_Y);
  float*    Dxy = (float*)(ws + OFF_DXY);
  float*    Dxx = (float*)(ws + OFF_DXX);
  float*    Dyy = (float*)(ws + OFF_DYY);
  float*    res = (float*)(ws + OFF_RES);

  // projection + L2 normalize (f16 output for the gram WMMAs)
  proj_norm_kernel<<<16 * 1024 / 16, 256, 0, stream>>>(feats, W, bias, Xh);
  proj_norm_kernel<<<16 * 768 / 16, 256, 0, stream>>>(tgts, W, bias, Yh);

  // squared-distance matrices via f16 WMMA grams, emitted diagonal-major
  gram_dist_kernel<<<dim3(64, 6, 16), 256, 0, stream>>>(Xh, Yh, Dxy, 1024, 768);
  gram_dist_kernel<<<dim3(64, 8, 16), 256, 0, stream>>>(Xh, Xh, Dxx, 1024, 1024);
  gram_dist_kernel<<<dim3(48, 6, 16), 256, 0, stream>>>(Yh, Yh, Dyy, 768, 768);

  // 48 concurrent skewed-wave soft-DTW scans
  sdtw_kernel<<<dim3(3, 16), 1024, 0, stream>>>(Dxy, Dxx, Dyy, res);

  // final scalar
  finish_kernel<<<1, 32, 0, stream>>>(res, (float*)d_out);
}
